// DeepSpeedMoEInference_50285477101613
// MI455X (gfx1250) — matrix operand order repaired and checked by reading, not compiled, not run on hardware
//
#include <hip/hip_runtime.h>

#define S_TOK 8192
#define MDIM  1024
#define HDIM  4096
#define NEXP  8
#define CAP   1024

#define BM 128
#define BN 128
#define BKK 64
#define LDAPAD 80   // LDS row stride in bf16 elems: 64 data + 16 pad (keeps 32B alignment)

typedef __bf16 bf16_t;
typedef __attribute__((ext_vector_type(16))) __bf16 v16bf;
typedef __attribute__((ext_vector_type(8)))  __bf16 v8bf;
typedef __attribute__((ext_vector_type(2)))  __bf16 v2bf;
typedef __attribute__((ext_vector_type(8)))  float  v8f;

// Native conversions: LLVM legalizes fptrunc f32->bf16 on gfx1250
// (v_cvt_pk_bf16_f32 / v_cvt_bf16_f32), RNE rounding.
__device__ __forceinline__ bf16_t f2bf(float f) { return (bf16_t)f; }

// pack two f32 -> two bf16 in one u32 (low = a, high = b)
__device__ __forceinline__ unsigned pk2bf(float a, float b) {
  v2bf r = { (bf16_t)a, (bf16_t)b };
  return __builtin_bit_cast(unsigned, r);
}

__device__ __forceinline__ v16bf cat8(v8bf lo, v8bf hi) {
  return __builtin_shufflevector(lo, hi, 0,1,2,3,4,5,6,7,8,9,10,11,12,13,14,15);
}

__device__ __forceinline__ float gelu_exact(float x) {
  return 0.5f * x * (1.0f + erff(x * 0.70710678118654752f));
}

// CDNA5 async copy: 16 bytes per lane, global -> LDS, tracked by ASYNCcnt.
__device__ __forceinline__ void async_copy_b128(unsigned lds_off, const void* gptr) {
  asm volatile("global_load_async_to_lds_b128 %0, %1, off"
               :: "v"(lds_off), "v"((unsigned long long)(uintptr_t)gptr)
               : "memory");
}
__device__ __forceinline__ void wait_async0() {
  asm volatile("s_wait_asynccnt 0x0" ::: "memory");
}

// ---------------------------------------------------------------------------
// Kernel 1: gating. One wave32 per token: logits = x[s]·wg, softmax, top-1.
// ---------------------------------------------------------------------------
__global__ __launch_bounds__(256) void gate_kernel(
    const float* __restrict__ x, const float* __restrict__ wg,
    int* __restrict__ eid, float* __restrict__ gate) {
  int lane = threadIdx.x & 31;
  int tok  = blockIdx.x * 8 + (threadIdx.x >> 5);
  const float* xr = x + (size_t)tok * MDIM;
  float acc[NEXP] = {};
  for (int m = lane; m < MDIM; m += 32) {
    float xv = xr[m];
    const float* wr = wg + m * NEXP;
#pragma unroll
    for (int e = 0; e < NEXP; ++e) acc[e] += xv * wr[e];
  }
#pragma unroll
  for (int off = 16; off > 0; off >>= 1) {
#pragma unroll
    for (int e = 0; e < NEXP; ++e) acc[e] += __shfl_down(acc[e], off, 32);
  }
  if (lane == 0) {
    float mx = acc[0]; int be = 0;
#pragma unroll
    for (int e = 1; e < NEXP; ++e) { if (acc[e] > mx) { mx = acc[e]; be = e; } }
    float sum = 0.f;
#pragma unroll
    for (int e = 0; e < NEXP; ++e) sum += __expf(acc[e] - mx);
    eid[tok]  = be;
    gate[tok] = 1.0f / sum;   // softmax value of the argmax logit
  }
}

// ---------------------------------------------------------------------------
// Kernel 2: sequential top-1 dispatch scan (single wave32, ballot/popc).
// slot[e*CAP + pos] = token s (or -1). Tokens past capacity are dropped.
// ---------------------------------------------------------------------------
__global__ void scan_kernel(const int* __restrict__ eid, int* __restrict__ slot) {
  int lane = threadIdx.x;
  for (int i = lane; i < NEXP * CAP; i += 32) slot[i] = -1;
  __threadfence();
  __syncthreads();
  int base[NEXP] = {};
  for (int c = 0; c < S_TOK / 32; ++c) {
    int s = c * 32 + lane;
    int e = eid[s];
#pragma unroll
    for (int q = 0; q < NEXP; ++q) {
      unsigned mm = (unsigned)__ballot(e == q);
      if (e == q) {
        int pos = base[q] + __popc(mm & ((1u << lane) - 1u));
        if (pos < CAP) slot[q * CAP + pos] = s;
      }
      base[q] += __popc(mm);
    }
  }
}

// ---------------------------------------------------------------------------
// Kernel 3: zero the output (dropped tokens contribute 0; d_out is poisoned).
// ---------------------------------------------------------------------------
__global__ __launch_bounds__(256) void zero_y(float4* __restrict__ y) {
  y[blockIdx.x * 256 + threadIdx.x] = make_float4(0.f, 0.f, 0.f, 0.f);
}

// ---------------------------------------------------------------------------
// Kernel 4: GEMM1: h = gelu( gather(x)[e] @ inter_w[e] + inter_b[e] ) -> bf16
// Tile 128x128, BK=64, 8 waves (2x4), each wave 4x2 16x16x32 bf16 WMMA tiles.
// ---------------------------------------------------------------------------
__global__ __launch_bounds__(256) void gemm1_kernel(
    const float* __restrict__ x, const float* __restrict__ inter_w,
    const float* __restrict__ inter_b, const int* __restrict__ slot,
    bf16_t* __restrict__ hbuf) {
  __shared__ bf16_t As[BM * LDAPAD];
  __shared__ bf16_t Bs[BN * LDAPAD];

  const int e  = blockIdx.z;
  const int tm = blockIdx.y;           // 0..7   (C tiles)
  const int tn = blockIdx.x;           // 0..31  (H tiles)
  const int t    = threadIdx.x;
  const int lane = t & 31;
  const int wv   = t >> 5;
  const int wm   = wv & 1;             // 2 wave-rows of 64
  const int wn   = wv >> 1;            // 4 wave-cols of 32
  const int lrow = lane & 15;
  const int hlf  = lane >> 4;

  const float* Bg    = inter_w + (size_t)e * MDIM * HDIM;  // [1024][4096]
  const int*   slotE = slot + e * CAP + tm * BM;

  v8f acc[4][2] = {};

  for (int k0 = 0; k0 < MDIM; k0 += BKK) {
    // --- A tile: gather token rows of x (branchless), f32 -> bf16 b64 stores ---
#pragma unroll
    for (int p = 0; p < 8; ++p) {
      int row = (t >> 4) + p * 16;
      int kc  = (t & 15) * 4;
      int s   = slotE[row];
      int sc  = s < 0 ? 0 : s;                      // clamp; select zeros below
      float4 v = *(const float4*)(x + (size_t)sc * MDIM + k0 + kc);
      if (s < 0) { v.x = 0.f; v.y = 0.f; v.z = 0.f; v.w = 0.f; }
      uint2 pk = make_uint2(pk2bf(v.x, v.y), pk2bf(v.z, v.w));
      *(uint2*)(As + row * LDAPAD + kc) = pk;
    }
    // --- B tile: [k][n] f32 -> LDS [n][k] bf16 (K contiguous per column) ---
#pragma unroll
    for (int p = 0; p < 8; ++p) {
      int k  = (t >> 5) + p * 8;
      int n4 = (t & 31) * 4;
      const float* src = Bg + (size_t)(k0 + k) * HDIM + tn * BN + n4;
      if (k0 + BKK < MDIM) __builtin_prefetch(src + (size_t)BKK * HDIM, 0, 0);
      float4 v = *(const float4*)src;
      unsigned p01 = pk2bf(v.x, v.y);
      unsigned p23 = pk2bf(v.z, v.w);
      *(unsigned short*)(Bs + (n4 + 0) * LDAPAD + k) = (unsigned short)p01;
      *(unsigned short*)(Bs + (n4 + 1) * LDAPAD + k) = (unsigned short)(p01 >> 16);
      *(unsigned short*)(Bs + (n4 + 2) * LDAPAD + k) = (unsigned short)p23;
      *(unsigned short*)(Bs + (n4 + 3) * LDAPAD + k) = (unsigned short)(p23 >> 16);
    }
    __syncthreads();
#pragma unroll
    for (int ks = 0; ks < 2; ++ks) {
      v16bf a[4], b[2];
#pragma unroll
      for (int i = 0; i < 4; ++i) {
        const bf16_t* rp = As + (wm * 64 + i * 16 + lrow) * LDAPAD + ks * 32;
        v8bf lo = *(const v8bf*)(rp + hlf * 16);       // K = half*16 .. +7
        v8bf hi = *(const v8bf*)(rp + hlf * 16 + 8);   // K = half*16+8 .. +7
        a[i] = cat8(lo, hi);
      }
#pragma unroll
      for (int j = 0; j < 2; ++j) {
        const bf16_t* cp = Bs + (wn * 32 + j * 16 + lrow) * LDAPAD + ks * 32 + hlf * 16;
        b[j] = *(const v16bf*)cp;                      // 16 contiguous K
      }
#pragma unroll
      for (int i = 0; i < 4; ++i)
#pragma unroll
        for (int j = 0; j < 2; ++j)
          acc[i][j] = __builtin_amdgcn_wmma_f32_16x16x32_bf16(
              false, a[i], false, b[j], (short)0, acc[i][j], false, false);
    }
    __syncthreads();
  }

  // --- epilogue: bias + exact GELU -> bf16 h ---
  const float* bias = inter_b + e * HDIM + tn * BN;
  bf16_t* out = hbuf + (size_t)e * CAP * HDIM + (size_t)(tm * BM) * HDIM + tn * BN;
#pragma unroll
  for (int i = 0; i < 4; ++i)
#pragma unroll
    for (int j = 0; j < 2; ++j)
#pragma unroll
      for (int v = 0; v < 8; ++v) {
        int r  = wm * 64 + i * 16 + hlf * 8 + v;       // C/D layout: row = vgpr + half*8
        int cN = wn * 32 + j * 16 + lrow;              // col = lane%16
        float val = acc[i][j][v] + bias[cN];
        out[(size_t)r * HDIM + cN] = f2bf(gelu_exact(val));
      }
}

// ---------------------------------------------------------------------------
// Kernel 5: GEMM2: exp_out = h[e] @ output_w[e] + output_b[e], fused combine:
// y[s] = gate[s] * exp_out[e, pos(s)]   (scatter via slot_to_token)
// A tile staged with CDNA5 async global->LDS copies (ASYNCcnt).
// ---------------------------------------------------------------------------
__global__ __launch_bounds__(256) void gemm2_kernel(
    const bf16_t* __restrict__ hbuf, const float* __restrict__ output_w,
    const float* __restrict__ output_b, const int* __restrict__ slot,
    const float* __restrict__ gate, float* __restrict__ y) {
  __shared__ bf16_t As[BM * LDAPAD];
  __shared__ bf16_t Bs[BN * LDAPAD];

  const int e  = blockIdx.z;
  const int tm = blockIdx.y;           // 0..7  (C tiles)
  const int tn = blockIdx.x;           // 0..7  (M tiles)
  const int t    = threadIdx.x;
  const int lane = t & 31;
  const int wv   = t >> 5;
  const int wm   = wv & 1;
  const int wn   = wv >> 1;
  const int lrow = lane & 15;
  const int hlf  = lane >> 4;

  const bf16_t* Ag = hbuf + (size_t)e * CAP * HDIM + (size_t)(tm * BM) * HDIM;
  const float*  Bg = output_w + (size_t)e * HDIM * MDIM;   // [4096][1024]

  v8f acc[4][2] = {};

  for (int k0 = 0; k0 < HDIM; k0 += BKK) {
    // --- A tile: bf16, async copy straight into LDS (no VGPR round-trip) ---
#pragma unroll
    for (int p = 0; p < 4; ++p) {
      int row = (t >> 3) + p * 32;
      int kc  = (t & 7) * 8;
      unsigned lds_off = (unsigned)(uintptr_t)(As + row * LDAPAD + kc);
      async_copy_b128(lds_off, Ag + (size_t)row * HDIM + k0 + kc);
    }
    // --- B tile: [k][n] f32 -> LDS [n][k] bf16 (overlaps the async copies) ---
#pragma unroll
    for (int p = 0; p < 8; ++p) {
      int k  = (t >> 5) + p * 8;
      int n4 = (t & 31) * 4;
      const float* src = Bg + (size_t)(k0 + k) * MDIM + tn * BN + n4;
      if (k0 + BKK < HDIM) __builtin_prefetch(src + (size_t)BKK * MDIM, 0, 0);
      float4 v = *(const float4*)src;
      unsigned p01 = pk2bf(v.x, v.y);
      unsigned p23 = pk2bf(v.z, v.w);
      *(unsigned short*)(Bs + (n4 + 0) * LDAPAD + k) = (unsigned short)p01;
      *(unsigned short*)(Bs + (n4 + 1) * LDAPAD + k) = (unsigned short)(p01 >> 16);
      *(unsigned short*)(Bs + (n4 + 2) * LDAPAD + k) = (unsigned short)p23;
      *(unsigned short*)(Bs + (n4 + 3) * LDAPAD + k) = (unsigned short)(p23 >> 16);
    }
    wait_async0();
    __syncthreads();
#pragma unroll
    for (int ks = 0; ks < 2; ++ks) {
      v16bf a[4], b[2];
#pragma unroll
      for (int i = 0; i < 4; ++i) {
        const bf16_t* rp = As + (wm * 64 + i * 16 + lrow) * LDAPAD + ks * 32;
        v8bf lo = *(const v8bf*)(rp + hlf * 16);
        v8bf hi = *(const v8bf*)(rp + hlf * 16 + 8);
        a[i] = cat8(lo, hi);
      }
#pragma unroll
      for (int j = 0; j < 2; ++j) {
        const bf16_t* cp = Bs + (wn * 32 + j * 16 + lrow) * LDAPAD + ks * 32 + hlf * 16;
        b[j] = *(const v16bf*)cp;
      }
#pragma unroll
      for (int i = 0; i < 4; ++i)
#pragma unroll
        for (int j = 0; j < 2; ++j)
          acc[i][j] = __builtin_amdgcn_wmma_f32_16x16x32_bf16(
              false, a[i], false, b[j], (short)0, acc[i][j], false, false);
    }
    __syncthreads();
  }

  // --- epilogue: bias + combine-scatter into y ---
  const float* bias  = output_b + e * MDIM + tn * BN;
  const int*   slotE = slot + e * CAP + tm * BM;
#pragma unroll
  for (int i = 0; i < 4; ++i)
#pragma unroll
    for (int v = 0; v < 8; ++v) {
      int r = wm * 64 + i * 16 + hlf * 8 + v;
      int s = slotE[r];
      if (s < 0) continue;                 // empty slot or dropped token
      float g = gate[s];
      float* yr = y + (size_t)s * MDIM + tn * BN;
#pragma unroll
      for (int j = 0; j < 2; ++j) {
        int cN = wn * 32 + j * 16 + lrow;
        yr[cN] = g * (acc[i][j][v] + bias[cN]);
      }
    }
}

// ---------------------------------------------------------------------------
extern "C" void kernel_launch(void* const* d_in, const int* in_sizes, int n_in,
                              void* d_out, int out_size, void* d_ws, size_t ws_size,
                              hipStream_t stream) {
  const float* x        = (const float*)d_in[0];   // [8192,1024]
  const float* wg       = (const float*)d_in[1];   // [1024,8]
  const float* inter_w  = (const float*)d_in[2];   // [8,1024,4096]
  const float* inter_b  = (const float*)d_in[3];   // [8,4096]
  const float* output_w = (const float*)d_in[4];   // [8,4096,1024]
  const float* output_b = (const float*)d_in[5];   // [8,1024]
  float* y = (float*)d_out;                        // [8192,1024]

  // workspace layout
  char* ws = (char*)d_ws;
  bf16_t* hbuf = (bf16_t*)ws;                                 // 8*1024*4096 bf16 = 64MB
  size_t hbytes = (size_t)NEXP * CAP * HDIM * sizeof(bf16_t);
  int*   slot  = (int*)(ws + hbytes);                         // 8192 ints
  int*   eidb  = slot + NEXP * CAP;                           // 8192 ints
  float* gateb = (float*)(eidb + S_TOK);                      // 8192 floats

  gate_kernel<<<S_TOK / 8, 256, 0, stream>>>(x, wg, eidb, gateb);
  scan_kernel<<<1, 32, 0, stream>>>(eidb, slot);
  zero_y<<<(S_TOK * MDIM / 4) / 256, 256, 0, stream>>>((float4*)y);
  gemm1_kernel<<<dim3(HDIM / BN, CAP / BM, NEXP), 256, 0, stream>>>(
      x, inter_w, inter_b, slot, hbuf);
  gemm2_kernel<<<dim3(MDIM / BN, CAP / BM, NEXP), 256, 0, stream>>>(
      hbuf, output_w, output_b, slot, gateb, y);
}
